// SwinTransformerBlock_66529043415700
// MI455X (gfx1250) — compile-verified, matrix operand order checked by reading, not run on hardware
//
#include <hip/hip_runtime.h>
#include <hip/hip_bf16.h>
#include <cstdint>

typedef __attribute__((ext_vector_type(16))) _Float16 v16h;
typedef __attribute__((ext_vector_type(8)))  float    v8f;

// ---------------- problem constants ----------------
#define Bn    32
#define Hh    56
#define Ww    56
#define WSz   7
#define SSz   3
#define NHd   8
#define Dm    256
#define HDd   32
#define Ls    49
#define NWn   64            // (56/7)^2
#define MT    100352        // Bn * Hh * Ww  == Bn*NWn*Ls
#define SCALEF 0.1767766952966369f   // 32^-0.5

// map a window-order token row -> spatial row index in (B, H*W) layout.
// shifted coord h' corresponds to original/final h = (h' + SS) mod H,
// which serves both the forward gather (LN1) and reverse scatter (proj).
__device__ __forceinline__ size_t win_to_spatial(int m) {
    int b  = m / (NWn * Ls);
    int r  = m % (NWn * Ls);
    int wi = r / Ls, l = r % Ls;
    int hp = (wi >> 3) * WSz + l / WSz;
    int wp = (wi & 7)  * WSz + l % WSz;
    int h = hp + SSz; if (h >= Hh) h -= Hh;
    int w = wp + SSz; if (w >= Ww) w -= Ww;
    return (size_t)b * (Hh * Ww) + (size_t)h * Ww + w;
}

// ---------------- weight conversion ----------------
__global__ __launch_bounds__(256) void f32_to_f16_kernel(const float* __restrict__ s,
                                                         _Float16* __restrict__ d, int n) {
    int i = blockIdx.x * 256 + threadIdx.x;
    if (i < n) d[i] = (_Float16)s[i];
}

// ---------------- LayerNorm (one wave per row, 8 rows/block) ----------------
template <int SHIFTED>
__global__ __launch_bounds__(256) void ln_kernel(const float* __restrict__ x,
                                                 const float* __restrict__ g,
                                                 const float* __restrict__ bta,
                                                 _Float16* __restrict__ out) {
    int wave = threadIdx.x >> 5, lane = threadIdx.x & 31;
    int row = blockIdx.x * 8 + wave;
    size_t srow = SHIFTED ? win_to_spatial(row) : (size_t)row;
    const float* xp = x + srow * Dm;
    float vals[8], s = 0.f, s2 = 0.f;
#pragma unroll
    for (int i = 0; i < 8; ++i) {
        float t = xp[lane + i * 32];
        vals[i] = t; s += t; s2 += t * t;
    }
#pragma unroll
    for (int o = 16; o >= 1; o >>= 1) {
        s  += __shfl_xor(s,  o, 32);
        s2 += __shfl_xor(s2, o, 32);
    }
    float mu  = s * (1.f / Dm);
    float var = s2 * (1.f / Dm) - mu * mu;
    float rs  = rsqrtf(var + 1e-5f);
    _Float16* op = out + (size_t)row * Dm;
#pragma unroll
    for (int i = 0; i < 8; ++i) {
        int c = lane + i * 32;
        op[c] = (_Float16)((vals[i] - mu) * rs * g[c] + bta[c]);
    }
}

// ---------------- generic WMMA GEMM: C = A(MxK) * B(KxN) ----------------
// block tile 128x64, 8 waves (4 along M x 2 along N), each wave 32x32 (2x2 wmma).
// Double-buffered LDS; B staged TRANSPOSED so both fragments load contiguously.
// EP: 0=qkv (bias, scale q cols, f16 out)  1=proj (bias + scattered residual, f32 out)
//     2=fc1 (bias + exact GELU, f16 out)   3=fc2 (bias + residual, f32 out)
template <int EP>
__global__ __launch_bounds__(256) void gemm_wmma_kernel(
    const _Float16* __restrict__ A, const _Float16* __restrict__ Bm,
    int N, int K,
    const float* __restrict__ bias,
    const float* __restrict__ resid,
    _Float16* __restrict__ outh,
    float* __restrict__ outf) {
    __shared__ _Float16 As[2][128][40];   // [buf][m][k], 32+8 pad
    __shared__ _Float16 Bs[2][64][40];    // [buf][n][k] (transposed), 32+8 pad

    const int tid = threadIdx.x;
    const int lane = tid & 31, wave = tid >> 5;
    const int wm = (wave & 3) * 32;
    const int wn = (wave >> 2) * 32;
    const int bm = blockIdx.y * 128;
    const int bn = blockIdx.x * 64;

    // staging indices
    const int ar0 = tid >> 2,        ac = (tid & 3) * 8;   // A: rows tid/4, tid/4+64
    const int br  = tid >> 3,        bc = (tid & 7) * 8;   // B: row kk+br, cols bn+bc..

    v8f acc[2][2] = {};
    uint4 ra[2], rb;

    auto load_tile = [&](int kk) {
        ra[0] = *(const uint4*)&A[(size_t)(bm + ar0) * K + kk + ac];
        ra[1] = *(const uint4*)&A[(size_t)(bm + ar0 + 64) * K + kk + ac];
        rb    = *(const uint4*)&Bm[(size_t)(kk + br) * N + bn + bc];
    };
    auto store_tile = [&](int buf) {
        *(uint4*)&As[buf][ar0][ac]      = ra[0];
        *(uint4*)&As[buf][ar0 + 64][ac] = ra[1];
        const _Float16* h = (const _Float16*)&rb;
#pragma unroll
        for (int i = 0; i < 8; ++i) Bs[buf][bc + i][br] = h[i];   // transpose scatter
    };

    const int nk = K >> 5;
    load_tile(0);
    for (int kt = 0; kt < nk; ++kt) {
        const int buf = kt & 1;
        store_tile(buf);
        __syncthreads();
        if (kt + 1 < nk) load_tile((kt + 1) << 5);   // issue next-tile global loads early

        const int fm = lane & 15, kh = (lane >> 4) * 16;
        v16h afr[2], bfr[2];
#pragma unroll
        for (int t = 0; t < 2; ++t) {
            const _Float16* p = &As[buf][wm + t * 16 + fm][kh];
#pragma unroll
            for (int i = 0; i < 16; ++i) afr[t][i] = p[i];
        }
#pragma unroll
        for (int t = 0; t < 2; ++t) {
            const _Float16* p = &Bs[buf][wn + t * 16 + fm][kh];
#pragma unroll
            for (int i = 0; i < 16; ++i) bfr[t][i] = p[i];
        }
#pragma unroll
        for (int i = 0; i < 2; ++i)
#pragma unroll
            for (int j = 0; j < 2; ++j)
                acc[i][j] = __builtin_amdgcn_wmma_f32_16x16x32_f16(
                    false, afr[i], false, bfr[j], (short)0, acc[i][j], false, false);
        // single barrier per iteration: next store targets the other buffer, and
        // this barrier (next iteration) guarantees reads of `buf` have drained
        // before it is overwritten two iterations later.
    }

    // epilogue: C/D layout — lanes 0-15: M=r, N=lane; lanes 16-31: M=r+8
    const int fn = lane & 15, mh = (lane >> 4) * 8;
#pragma unroll
    for (int i = 0; i < 2; ++i)
#pragma unroll
        for (int j = 0; j < 2; ++j)
#pragma unroll
            for (int r = 0; r < 8; ++r) {
                int grow = bm + wm + i * 16 + mh + r;
                int gcol = bn + wn + j * 16 + fn;
                float v = acc[i][j][r] + bias[gcol];
                if (EP == 0) {                       // qkv: scale q block
                    if (gcol < Dm) v *= SCALEF;
                    outh[(size_t)grow * N + gcol] = (_Float16)v;
                } else if (EP == 1) {                // proj: unshift + residual
                    size_t drow = win_to_spatial(grow);
                    outf[drow * Dm + gcol] = v + resid[drow * Dm + gcol];
                } else if (EP == 2) {                // fc1: exact GELU
                    float gl = 0.5f * v * (1.f + erff(v * 0.70710678118654752f));
                    outh[(size_t)grow * N + gcol] = (_Float16)gl;
                } else {                             // fc2: residual, final out
                    outf[(size_t)grow * Dm + gcol] = v + resid[(size_t)grow * Dm + gcol];
                }
            }
}

// ---------------- attention: one block per (window, head) ----------------
__device__ __forceinline__ float attn_val(float Sv, int i, int j, int head, int wimg,
                                          const float* __restrict__ rel_bias) {
    if (j >= Ls) return -1e9f;       // padded keys
    if (i >= Ls) return 0.f;         // padded query rows: uniform, discarded later
    int hi = i / WSz, wi = i % WSz, hj = j / WSz, wj = j % WSz;
    int idx = (hi - hj + WSz - 1) * (2 * WSz - 1) + (wi - wj + WSz - 1);
    float b = rel_bias[idx * NHd + head];
    int wr = wimg >> 3, wc = wimg & 7;
    int hpi = wr * WSz + hi, wpi = wc * WSz + wi;
    int hpj = wr * WSz + hj, wpj = wc * WSz + wj;
    auto reg = [](int hp, int wp) {
        int rh = hp < (Hh - WSz) ? 0 : (hp < (Hh - SSz) ? 1 : 2);
        int rw = wp < (Ww - WSz) ? 0 : (wp < (Ww - SSz) ? 1 : 2);
        return rh * 3 + rw;
    };
    float mask = (reg(hpi, wpi) == reg(hpj, wpj)) ? -100.f : 0.f;   // faithful to source
    return Sv + b + mask;
}

__global__ __launch_bounds__(256) void attn_kernel(const _Float16* __restrict__ qkv,
                                                   const float* __restrict__ rel_bias,
                                                   _Float16* __restrict__ outh) {
    __shared__ _Float16 Qs[64][40], Ks[64][40];
    __shared__ _Float16 Vt[32][72];          // V transposed: [d][k], 64+8 pad
    __shared__ float    S[64][65];
    __shared__ _Float16 P[64][72];

    const int tid = threadIdx.x, lane = tid & 31, wave = tid >> 5;
    const int head = blockIdx.x & 7;
    const int win  = blockIdx.x >> 3;   // b*NW + wi
    const int wimg = win & (NWn - 1);

    // load Q,K (row-major, zero-pad to 64 rows) and V transposed
    {
        int r = tid >> 2, c = (tid & 3) * 8;
        uint4 q4 = {0, 0, 0, 0}, k4 = q4, v4 = q4;
        if (r < Ls) {
            size_t rb = ((size_t)win * Ls + r) * (3 * Dm) + head * HDd + c;
            q4 = *(const uint4*)&qkv[rb];
            k4 = *(const uint4*)&qkv[rb + Dm];
            v4 = *(const uint4*)&qkv[rb + 2 * Dm];
        }
        *(uint4*)&Qs[r][c] = q4;
        *(uint4*)&Ks[r][c] = k4;
        const _Float16* vh = (const _Float16*)&v4;
#pragma unroll
        for (int i = 0; i < 8; ++i) Vt[c + i][r] = vh[i];   // transpose scatter
    }
    __syncthreads();

    const int fm = lane & 15, kh = (lane >> 4) * 16;
    const int fn = lane & 15, mh = (lane >> 4) * 8;

    // S = Q @ K^T  (64x64, 16 tiles of 16x16, 2 per wave; K-dim = HD = 32 exactly)
#pragma unroll
    for (int t = 0; t < 2; ++t) {
        int tile = wave * 2 + t, ti = tile >> 2, tj = tile & 3;
        v16h a, b;
        {
            const _Float16* p = &Qs[ti * 16 + fm][kh];
#pragma unroll
            for (int i = 0; i < 16; ++i) a[i] = p[i];
        }
        {
            const _Float16* p = &Ks[tj * 16 + fm][kh];   // B = K^T -> lane holds a K row
#pragma unroll
            for (int i = 0; i < 16; ++i) b[i] = p[i];
        }
        v8f acc = {};
        acc = __builtin_amdgcn_wmma_f32_16x16x32_f16(false, a, false, b, (short)0, acc,
                                                     false, false);
#pragma unroll
        for (int r = 0; r < 8; ++r) S[ti * 16 + mh + r][tj * 16 + fn] = acc[r];
    }
    __syncthreads();

    // softmax: 4 threads per row (16 cols each), quad reduction via shfl_xor
    {
        int i = tid >> 2, sub = tid & 3;
        float mx = -1e30f;
#pragma unroll
        for (int jj = 0; jj < 16; ++jj) {
            int j = sub * 16 + jj;
            mx = fmaxf(mx, attn_val(S[i][j], i, j, head, wimg, rel_bias));
        }
        mx = fmaxf(mx, __shfl_xor(mx, 1, 32));
        mx = fmaxf(mx, __shfl_xor(mx, 2, 32));
        float sum = 0.f;
#pragma unroll
        for (int jj = 0; jj < 16; ++jj) {
            int j = sub * 16 + jj;
            float e = expf(attn_val(S[i][j], i, j, head, wimg, rel_bias) - mx);
            S[i][j] = e; sum += e;
        }
        sum += __shfl_xor(sum, 1, 32);
        sum += __shfl_xor(sum, 2, 32);
        float inv = 1.f / sum;
#pragma unroll
        for (int jj = 0; jj < 16; ++jj) {
            int j = sub * 16 + jj;
            P[i][j] = (_Float16)(S[i][j] * inv);
        }
    }
    __syncthreads();

    // O = P @ V  (64x32, 8 tiles, one per wave; K-dim 64 = two wmma steps)
    {
        int ti = wave >> 1, tj = wave & 1;
        v8f acc = {};
#pragma unroll
        for (int ks = 0; ks < 2; ++ks) {
            v16h a, b;
            const _Float16* p = &P[ti * 16 + fm][ks * 32 + kh];
#pragma unroll
            for (int i = 0; i < 16; ++i) a[i] = p[i];
            const _Float16* q = &Vt[tj * 16 + fm][ks * 32 + kh];   // contiguous via Vt
#pragma unroll
            for (int i = 0; i < 16; ++i) b[i] = q[i];
            acc = __builtin_amdgcn_wmma_f32_16x16x32_f16(false, a, false, b, (short)0, acc,
                                                         false, false);
        }
#pragma unroll
        for (int r = 0; r < 8; ++r) {
            int m = ti * 16 + mh + r;
            if (m < Ls)
                outh[((size_t)win * Ls + m) * Dm + head * HDd + tj * 16 + fn] =
                    (_Float16)acc[r];
        }
    }
}

// ---------------- host launcher ----------------
extern "C" void kernel_launch(void* const* d_in, const int* in_sizes, int n_in,
                              void* d_out, int out_size, void* d_ws, size_t ws_size,
                              hipStream_t stream) {
    const float* x       = (const float*)d_in[0];
    const float* g1      = (const float*)d_in[1];
    const float* beta1   = (const float*)d_in[2];
    const float* w_qkv   = (const float*)d_in[3];
    const float* b_qkv   = (const float*)d_in[4];
    const float* relb    = (const float*)d_in[5];
    const float* w_proj  = (const float*)d_in[6];
    const float* b_proj  = (const float*)d_in[7];
    const float* g2      = (const float*)d_in[8];
    const float* beta2   = (const float*)d_in[9];
    const float* w_fc1   = (const float*)d_in[10];
    const float* b_fc1   = (const float*)d_in[11];
    const float* w_fc2   = (const float*)d_in[12];
    const float* b_fc2   = (const float*)d_in[13];
    float* out = (float*)d_out;

    char* ws = (char*)d_ws;
    size_t off = 0;
    auto take = [&](size_t bytes) -> char* {
        char* p = ws + off;
        off = (off + bytes + 255) & ~(size_t)255;
        return p;
    };
    _Float16* wqkv_h = (_Float16*)take((size_t)Dm * 3 * Dm * 2);
    _Float16* wprj_h = (_Float16*)take((size_t)Dm * Dm * 2);
    _Float16* wfc1_h = (_Float16*)take((size_t)Dm * 4 * Dm * 2);
    _Float16* wfc2_h = (_Float16*)take((size_t)4 * Dm * Dm * 2);
    _Float16* bufA   = (_Float16*)take((size_t)MT * Dm * 2);       // a0 / attn_out / a2
    _Float16* bufB   = (_Float16*)take((size_t)MT * 4 * Dm * 2);   // qkv (768 cols) / h1 (1024 cols)
    float*    xnew   = (float*)take((size_t)MT * Dm * 4);

    // 0) weights -> f16
    auto cvt = [&](const float* s, _Float16* d, int n) {
        f32_to_f16_kernel<<<(n + 255) / 256, 256, 0, stream>>>(s, d, n);
    };
    cvt(w_qkv, wqkv_h, Dm * 3 * Dm);
    cvt(w_proj, wprj_h, Dm * Dm);
    cvt(w_fc1, wfc1_h, Dm * 4 * Dm);
    cvt(w_fc2, wfc2_h, 4 * Dm * Dm);

    // 1) LN1 + shift + window partition -> bufA (f16, window order)
    ln_kernel<1><<<MT / 8, 256, 0, stream>>>(x, g1, beta1, bufA);

    // 2) QKV GEMM (MT x 256) x (256 x 768) -> bufB f16, q pre-scaled
    gemm_wmma_kernel<0><<<dim3(3 * Dm / 64, MT / 128), 256, 0, stream>>>(
        bufA, wqkv_h, 3 * Dm, Dm, b_qkv, nullptr, bufB, nullptr);

    // 3) windowed attention -> bufA f16 (window order, L x D per window)
    attn_kernel<<<Bn * NWn * NHd, 256, 0, stream>>>(bufB, relb, bufA);

    // 4) proj GEMM + window-reverse/unshift + residual -> xnew f32
    gemm_wmma_kernel<1><<<dim3(Dm / 64, MT / 128), 256, 0, stream>>>(
        bufA, wprj_h, Dm, Dm, b_proj, x, nullptr, xnew);

    // 5) LN2 -> bufA f16
    ln_kernel<0><<<MT / 8, 256, 0, stream>>>(xnew, g2, beta2, bufA);

    // 6) FC1 GEMM + GELU -> bufB f16 (MT x 1024)
    gemm_wmma_kernel<2><<<dim3(4 * Dm / 64, MT / 128), 256, 0, stream>>>(
        bufA, wfc1_h, 4 * Dm, Dm, b_fc1, nullptr, bufB, nullptr);

    // 7) FC2 GEMM + residual -> d_out f32
    gemm_wmma_kernel<3><<<dim3(Dm / 64, MT / 128), 256, 0, stream>>>(
        bufB, wfc2_h, Dm, 4 * Dm, b_fc2, xnew, nullptr, out);
}
